// tree_classifier_16441134809397
// MI455X (gfx1250) — matrix operand order repaired and checked by reading, not compiled, not run on hardware
//
#include <hip/hip_runtime.h>
#include <hip/hip_bf16.h>

// ---------------------------------------------------------------------------
// Tree-GRU (ChildSum, complete binary forest) for MI455X / gfx1250.
//
//  * Fuse W-GEMM + U-GEMM per tree level: pre = [x | childsum(h)] @ [W_uo|U_uo]^T
//    ('r' gate of the reference is dead code -> its 128 rows are skipped).
//  * All matmul work in v_wmma_f32_16x16x32_bf16; h kept bf16 in workspace.
//  * x read exactly once from HBM (268 MB); h traffic 2x134 MB bf16.
//  * Weights pre-swizzled to WMMA-fragment order; per level-kernel block the
//    B k-slices are double-buffered into LDS with global_load_async_to_lds_b128
//    (ASYNCcnt) so slice kb+1 streams while slice kb feeds the WMMAs.
//  * A tile staged in fragment-native layout: every fragment read is a
//    conflict-free 32B/lane ds_load_b128 pair.
//  * B fragment registers double-buffered in-wave: loads for tile nt+1 issue
//    before the WMMA for tile nt, so LDS latency overlaps matrix issue.
//
// Workspace layout (needs ~134.4 MB):
//   [0, 128KB)        : bf16 weights in fragment order, kb-major (8 x 16KB slices)
//   [128KB, +134.2MB) : h_all bf16, heap node layout [B][8191][128]
// ---------------------------------------------------------------------------

typedef __attribute__((ext_vector_type(16))) __bf16        v16bf;
typedef __attribute__((ext_vector_type(8)))  float         v8f;
typedef __attribute__((ext_vector_type(8)))  unsigned int  v8u;

#define B_TREES   64
#define TDEPTH    12
#define NNODES    8191            // 2^(DEPTH+1)-1
#define HS        128
#define XS        128
#define WFRAG_ELEMS (8 * 16 * 32 * 16)   // 8 k-blocks x 16 n-tiles x 32 lanes x 16 bf16
#define SLICE_USH   (16 * 32 * 16)       // one k-block slice: 8192 ushort = 16 KB

__device__ __forceinline__ unsigned short f2bf(float f) {
    union { float f; unsigned int u; } v; v.f = f;
    unsigned int r = v.u + 0x7FFFu + ((v.u >> 16) & 1u);   // round-to-nearest-even
    return (unsigned short)(r >> 16);
}
__device__ __forceinline__ float bf2f(unsigned short h) {
    union { unsigned int u; float f; } v; v.u = ((unsigned int)h) << 16;
    return v.f;
}

// LDS offset of a generic __shared__ pointer = low 32 bits of the flat address
__device__ __forceinline__ unsigned lds_offset(const void* p) {
    return (unsigned)(unsigned long long)p;
}

// gfx1250 async copy global -> LDS, 16B per lane, tracked by ASYNCcnt
__device__ __forceinline__ void async_g2l_b128(unsigned ldsoff, const void* gaddr) {
    asm volatile("global_load_async_to_lds_b128 %0, %1, off"
                 :: "v"(ldsoff), "v"(gaddr) : "memory");
}
__device__ __forceinline__ void wait_asynccnt0() {
    asm volatile("s_wait_asynccnt 0" ::: "memory");
}

// A-fragment slot for element (row 0..63, k 0..255) in fragment-native LDS:
//   fragment id = (row/16)*8 + k/32 ; lane = ((k>>3)&1)*16 + row%16
//   val within lane's 16 bf16 = ((k>>4)&1)*8 + k%8
__device__ __forceinline__ unsigned short* a_slot(unsigned short* ldsA, int row, int k) {
    int fragid = ((row >> 4) << 3) + (k >> 5);
    int lane   = (((k >> 3) & 1) << 4) + (row & 15);
    int val    = (((k >> 4) & 1) << 3) + (k & 7);
    return &ldsA[(((fragid << 5) + lane) << 4) + val];
}

// ---------------------------------------------------------------------------
// Swizzle useful weight rows (u,o gates: rows 128..383 of W_ruo / U_ruo) into
// WMMA B-fragment order, kb-outermost so each k-block slice is contiguous:
//   e = ((kb*16 + nt)*32 + lane)*16 + val
//   column N = nt*16 + lane%16 ; k = kb*32 + (val/8)*16 + (lane/16)*8 + val%8
//   k <  128 -> W_ruo[(128+N)*128 + k]
//   k >= 128 -> U_ruo[(128+N)*128 + (k-128)]
// ---------------------------------------------------------------------------
__global__ void prep_weights_kernel(const float* __restrict__ W,
                                    const float* __restrict__ U,
                                    unsigned short* __restrict__ wfrag) {
    for (int e = threadIdx.x; e < WFRAG_ELEMS; e += blockDim.x) {
        int val   = e & 15;
        int lane  = (e >> 4) & 31;
        int nt    = (e >> 9) & 15;
        int kb    = e >> 13;
        int ncol  = (nt << 4) + (lane & 15);
        int k     = (kb << 5) + ((val >> 3) << 4) + ((lane >> 4) << 3) + (val & 7);
        float s = (k < 128) ? W[(128 + ncol) * 128 + k]
                            : U[(128 + ncol) * 128 + (k - 128)];
        wfrag[e] = f2bf(s);
    }
}

// ---------------------------------------------------------------------------
// One tree level. Grid = 2^lvl blocks of 128 threads (4 waves).
// Block owns 64 rows x 256 out-cols; each wave owns 16 rows x 256 cols
// (16 f32 accumulator tiles: nt 0..7 = u-gate pre, nt 8..15 = o-gate pre).
// ---------------------------------------------------------------------------
__global__ __launch_bounds__(128)
void tree_level_kernel(const float* __restrict__ x,
                       unsigned short* __restrict__ h_all,
                       const unsigned short* __restrict__ wfrag,
                       const float* __restrict__ bias,
                       int lvl, int is_leaf) {
    __shared__ unsigned short ldsA[32 * 32 * 16];     // 32 KB: A fragments (4 rowtiles x 8 kb)
    __shared__ unsigned short ldsB[2][SLICE_USH];     // 32 KB: double-buffered B k-slices

    const int tid       = threadIdx.x;
    const int size      = 1 << lvl;
    const int start     = size - 1;            // first node of this level
    const int nextstart = 2 * size - 1;        // first node of child level
    const int m0        = blockIdx.x * 64;     // first global row of this block
    const int kblocks   = is_leaf ? 4 : 8;     // leaves: only the x half of K

    // ---- kick off async staging of B slice 0 ----------------------------
    {
        const char* g0 = (const char*)wfrag;                // slice kb=0
        unsigned l0 = lds_offset(&ldsB[0][0]);
        #pragma unroll
        for (int i = 0; i < 8; ++i) {
            int c = tid + i * 128;                          // 16B chunk id, 0..1023
            async_g2l_b128(l0 + c * 16, g0 + c * 16);
        }
    }

    // ---- stage A tile (x -> bf16, childsum(h) -> bf16) in fragment order -
    for (int idx = tid; idx < 64 * 32; idx += 128) {
        int row   = idx >> 5;
        int kbase = (idx & 31) << 2;                        // 4 consecutive k
        int m  = m0 + row;
        int bt = m >> lvl;                                  // tree index
        int ni = m & (size - 1);                            // node-in-level
        float4 xv = *(const float4*)(x + ((long long)bt * NNODES + (start + ni)) * XS + kbase);
        unsigned short* dx = a_slot(ldsA, row, kbase);      // 4 consecutive vals
        dx[0] = f2bf(xv.x); dx[1] = f2bf(xv.y); dx[2] = f2bf(xv.z); dx[3] = f2bf(xv.w);
        if (!is_leaf) {
            long long c0 = (long long)bt * NNODES + (nextstart + 2 * ni);
            ushort4 ha = *(const ushort4*)(h_all + c0 * HS + kbase);
            ushort4 hb = *(const ushort4*)(h_all + (c0 + 1) * HS + kbase);
            unsigned short* dh = a_slot(ldsA, row, 128 + kbase);
            dh[0] = f2bf(bf2f(ha.x) + bf2f(hb.x));
            dh[1] = f2bf(bf2f(ha.y) + bf2f(hb.y));
            dh[2] = f2bf(bf2f(ha.z) + bf2f(hb.z));
            dh[3] = f2bf(bf2f(ha.w) + bf2f(hb.w));
        }
    }
    __syncthreads();                                        // A tile visible

    // ---- WMMA main loop: slice kb feeds WMMAs while slice kb+1 streams --
    const int wave    = tid >> 5;
    const int lane    = tid & 31;
    const int rowtile = wave;                               // 16-row tile per wave

    v8f acc[16] = {};                                       // nt 0..7 = u, 8..15 = o

    for (int kb = 0; kb < kblocks; ++kb) {
        wait_asynccnt0();                                   // this wave's share landed
        __syncthreads();                                    // whole slice kb visible
        if (kb + 1 < kblocks) {                             // stream next slice
            const char* gn = (const char*)wfrag + (size_t)(kb + 1) * (SLICE_USH * 2);
            unsigned ln = lds_offset(&ldsB[(kb + 1) & 1][0]);
            #pragma unroll
            for (int i = 0; i < 8; ++i) {
                int c = tid + i * 128;
                async_g2l_b128(ln + c * 16, gn + c * 16);
            }
        }
        const unsigned short* ldsBc = &ldsB[kb & 1][0];
        v8u araw = *(const v8u*)&ldsA[((((rowtile << 3) + kb) << 5) + lane) << 4];
        v16bf afrag = __builtin_bit_cast(v16bf, araw);

        // B fragments double-buffered in registers: load nt+1 before wmma nt
        v8u braw[2];
        braw[0] = *(const v8u*)&ldsBc[(lane) << 4];
        #pragma unroll
        for (int nt = 0; nt < 16; ++nt) {
            if (nt + 1 < 16)
                braw[(nt + 1) & 1] = *(const v8u*)&ldsBc[(((nt + 1) << 5) + lane) << 4];
            v16bf bfrag = __builtin_bit_cast(v16bf, braw[nt & 1]);
            acc[nt] = __builtin_amdgcn_wmma_f32_16x16x32_bf16(
                false, afrag, false, bfrag, (short)0, acc[nt], false, false);
        }
    }

    // ---- epilogue: u=sigmoid, o=tanh, h = o*u + (1-u)*h_tild ------------
    // C layout: VGPR j -> M = j + (lane/16)*8 ; N = lane%16.
    const int lm    = lane & 15;
    const int jbase = (lane >> 4) << 3;
    #pragma unroll
    for (int nt = 0; nt < 8; ++nt) {
        int k  = (nt << 4) + lm;
        float bu = bias[128 + k];
        float bo = bias[256 + k];
        #pragma unroll
        for (int j = 0; j < 8; ++j) {
            int row = rowtile * 16 + jbase + j;
            float u  = 1.0f / (1.0f + __expf(-(acc[nt][j] + bu)));
            float o  = tanhf(acc[nt + 8][j] + bo);
            float ht = is_leaf ? 0.0f : bf2f(*a_slot(ldsA, row, 128 + k));
            float h  = o * u + (1.0f - u) * ht;
            int m  = m0 + row;
            int bt = m >> lvl;
            int ni = m & (size - 1);
            h_all[((long long)bt * NNODES + (start + ni)) * HS + k] = f2bf(h);
        }
    }
}

// ---------------------------------------------------------------------------
// out[b][c] = fc_b[c] + sum_k h_root[b][k] * fc_w[c][k]   (64 x 2)
// ---------------------------------------------------------------------------
__global__ void final_fc_kernel(const unsigned short* __restrict__ h_all,
                                const int* __restrict__ root_idx,
                                const float* __restrict__ fcw,
                                const float* __restrict__ fcb,
                                float* __restrict__ out) {
    int t = threadIdx.x;
    if (t >= B_TREES * 2) return;
    int bt = t >> 1, c = t & 1;
    long long node = root_idx[bt];                 // root level starts at node 0
    const unsigned short* h = h_all + ((long long)bt * NNODES + node) * HS;
    float s = fcb[c];
    for (int k = 0; k < HS; ++k) s += bf2f(h[k]) * fcw[c * HS + k];
    out[t] = s;
}

// ---------------------------------------------------------------------------
extern "C" void kernel_launch(void* const* d_in, const int* in_sizes, int n_in,
                              void* d_out, int out_size, void* d_ws, size_t ws_size,
                              hipStream_t stream) {
    const float* x    = (const float*)d_in[0];
    const int*   ri   = (const int*)  d_in[1];
    const float* W    = (const float*)d_in[2];
    const float* U    = (const float*)d_in[3];
    const float* bias = (const float*)d_in[4];
    const float* fcw  = (const float*)d_in[5];
    const float* fcb  = (const float*)d_in[6];

    unsigned short* wfrag = (unsigned short*)d_ws;                              // 128 KB
    unsigned short* h_all = (unsigned short*)((char*)d_ws + (WFRAG_ELEMS * 2)); // 134.2 MB

    prep_weights_kernel<<<1, 256, 0, stream>>>(W, U, wfrag);

    for (int l = TDEPTH; l >= 0; --l) {
        tree_level_kernel<<<(1 << l), 128, 0, stream>>>(
            x, h_all, wfrag, bias, l, (l == TDEPTH) ? 1 : 0);
    }

    final_fc_kernel<<<1, 128, 0, stream>>>(h_all, ri, fcw, fcb, (float*)d_out);
}
